// BaiChuanAttention_21534966022787
// MI455X (gfx1250) — matrix operand order repaired and checked by reading, not compiled, not run on hardware
//
#include <hip/hip_runtime.h>
#include <hip/hip_bf16.h>

// ---------------------------------------------------------------------------
// Types for CDNA5 WMMA (wave32): bf16 16x16x32, fp32 accumulate
// ---------------------------------------------------------------------------
typedef __attribute__((ext_vector_type(16))) __bf16        v16bf;
typedef __attribute__((ext_vector_type(8)))  float         v8f;
typedef __attribute__((ext_vector_type(8)))  unsigned int  v8u;
typedef __attribute__((ext_vector_type(4)))  unsigned int  v4u;
typedef __attribute__((ext_vector_type(8)))  int           v8i;
typedef __attribute__((ext_vector_type(4)))  int           v4i;

#if defined(__has_builtin)
#  if __has_builtin(__builtin_amdgcn_tensor_load_to_lds)
#    define HAVE_TDM 1
#  else
#    define HAVE_TDM 0
#  endif
#else
#  define HAVE_TDM 0
#endif

__device__ __forceinline__ unsigned short f32_to_bf16(float f) {
    unsigned int u = __builtin_bit_cast(unsigned int, f);
    unsigned int r = (u + 0x7FFFu + ((u >> 16) & 1u)) >> 16;   // RNE
    return (unsigned short)r;
}
__device__ __forceinline__ float bf16_to_f32(unsigned short h) {
    unsigned int u = ((unsigned int)h) << 16;
    return __builtin_bit_cast(float, u);
}

// Build a 16-element bf16 fragment (8 VGPRs) from two 16B chunks.
__device__ __forceinline__ v16bf mk16(uint4 lo, uint4 hi) {
    v8u r;
    r[0] = lo.x; r[1] = lo.y; r[2] = lo.z; r[3] = lo.w;
    r[4] = hi.x; r[5] = hi.y; r[6] = hi.z; r[7] = hi.w;
    return __builtin_bit_cast(v16bf, r);
}

__device__ __forceinline__ v8f wmma_bf16(v16bf a, v16bf b, v8f c) {
    return __builtin_amdgcn_wmma_f32_16x16x32_bf16(false, a, false, b,
                                                   (short)0, c, false, false);
}

// Intra-wave LDS visibility: drain DS counter, pin program order.
__device__ __forceinline__ void wave_lds_sync() {
    asm volatile("s_wait_dscnt 0x0" ::: "memory");
    __builtin_amdgcn_wave_barrier();
}

// Low 32 bits of a generic pointer into LDS == workgroup LDS byte offset
// (ISA: aperture mapping; the LDS address is the low 32 address bits).
__device__ __forceinline__ unsigned lds_off32(const void* p) {
    return (unsigned)(unsigned long long)p;
}

#if HAVE_TDM
// ---------------------------------------------------------------------------
// Tensor Data Mover: DMA a rows x cols bf16 tile (row stride in elements)
// from global memory into LDS, packed row-major (cols per row, compacted).
// Descriptor per CDNA5 ISA ch.8: group0 {count=1, lds_addr, global_addr,
// type=2}, group1 {data_size=2B, tensor dims == tile dims (always in-bounds),
// tile dims, dim0 stride}. One issue per wave; tracked by TENSORcnt.
// This toolchain exposes the 6-arg builtin (extra int32x8 group, then cpol).
// ---------------------------------------------------------------------------
__device__ __forceinline__ void tdm_load_2d(unsigned lds_off, const void* gsrc,
                                            int rows, int cols,
                                            int row_stride_elems) {
    unsigned long long ga = (unsigned long long)gsrc;
    v4u g0 = {0u, 0u, 0u, 0u};
    g0[0] = 1u;                                   // count=1 (valid user desc)
    g0[1] = lds_off;                              // lds_addr
    g0[2] = (unsigned)ga;                         // global_addr[31:0]
    g0[3] = (unsigned)((ga >> 32) & 0x01FFFFFFull) | (2u << 30); // addr|type=2
    v8i g1;
    g1[0] = (int)(1u << 16);                      // data_size = 2 bytes
    g1[1] = (int)(((unsigned)cols & 0xFFFFu) << 16);             // dim0 lo16
    g1[2] = (int)((((unsigned)cols >> 16) & 0xFFFFu) |
                  (((unsigned)rows & 0xFFFFu) << 16));           // dim0hi|dim1lo
    g1[3] = (int)((((unsigned)rows >> 16) & 0xFFFFu) |
                  (((unsigned)cols & 0xFFFFu) << 16));           // dim1hi|tile0
    g1[4] = (int)((unsigned)rows & 0xFFFFu);      // tile_dim1 (tile_dim2 = 0)
    g1[5] = row_stride_elems;                     // tensor_dim0_stride[31:0]
    g1[6] = 0;
    g1[7] = 0;
    v4i g2 = {0, 0, 0, 0}, g3 = {0, 0, 0, 0};
    v8i g4 = {0, 0, 0, 0, 0, 0, 0, 0};
    __builtin_amdgcn_tensor_load_to_lds(g0, g1, g2, g3, g4, 0);
}
#endif

// ---------------------------------------------------------------------------
// fp32 -> bf16 conversion (4 elems/thread, vectorized)
// ---------------------------------------------------------------------------
__global__ __launch_bounds__(256) void cvt_f32_bf16_kernel(
        const float* __restrict__ in, unsigned short* __restrict__ out, long n) {
    long i = ((long)blockIdx.x * blockDim.x + threadIdx.x) * 4;
    if (i + 3 < n) {
        float4 f = *(const float4*)(in + i);
        ushort4 o;
        o.x = f32_to_bf16(f.x); o.y = f32_to_bf16(f.y);
        o.z = f32_to_bf16(f.z); o.w = f32_to_bf16(f.w);
        *(ushort4*)(out + i) = o;
    }
}

// ---------------------------------------------------------------------------
// Tiled bf16 GEMM: C[M,N] = A[M,K] * B[K,N], fp32 accumulate.
// Block 256 threads = 8 waves, C tile 128x128, BK=32. Double-buffered:
//   - A tile (pure 2D copy) streamed by TDM into LDS (wave 0 issues).
//   - B tile register-staged, scattered TRANSPOSED into LDS *after* the
//     fragment ds_loads so the in-order DS queue never stalls WMMA operands.
// Fragment layouts follow the ISA 16-bit A/B VGPR tables exactly.
// ---------------------------------------------------------------------------
template <bool OUT_BF16>
__global__ __launch_bounds__(256) void gemm_bf16_kernel(
        const unsigned short* __restrict__ A,   // [M,K] bf16
        const unsigned short* __restrict__ Bm,  // [K,N] bf16
        void* __restrict__ C, int M, int N, int K) {
    __shared__ unsigned short a_lds[2][128 * 32];   // [buf][row][k]
    __shared__ unsigned short b_lds[2][128 * 32];   // [buf][col][k] (transposed)

    const int tid  = threadIdx.x;
    const int lane = tid & 31;
    const int wave = tid >> 5;
    const int hl   = (lane >> 4) & 1;
    const int lc   = lane & 15;
    const int m0   = blockIdx.y * 128;
    const int n0   = blockIdx.x * 128;

    const v8f vzero = {0.f, 0.f, 0.f, 0.f, 0.f, 0.f, 0.f, 0.f};
    v8f acc[8];
#pragma unroll
    for (int g = 0; g < 8; ++g) acc[g] = vzero;

    // ---- prologue: stage k0 = 0 into buffer 0 ----
#if HAVE_TDM
    if (wave == 0)
        tdm_load_2d(lds_off32(&a_lds[0][0]), A + (size_t)m0 * K, 128, 32, K);
#else
#pragma unroll
    for (int i = 0; i < 2; ++i) {
        int idx = tid + i * 256;
        int row = idx >> 2, kq = (idx & 3) << 3;
        uint4 d = *(const uint4*)(A + (size_t)(m0 + row) * K + kq);
        *(uint4*)(&a_lds[0][row * 32 + kq]) = d;
    }
#endif
    {
        uint4 r[2];
#pragma unroll
        for (int i = 0; i < 2; ++i) {
            int idx = tid + i * 256;
            int kk = idx >> 4, nq = (idx & 15) << 3;
            r[i] = *(const uint4*)(Bm + (size_t)kk * N + n0 + nq);
        }
#pragma unroll
        for (int i = 0; i < 2; ++i) {
            int idx = tid + i * 256;
            int kk = idx >> 4, nq = (idx & 15) << 3;
            const unsigned short* e = (const unsigned short*)&r[i];
#pragma unroll
            for (int j = 0; j < 8; ++j) b_lds[0][(nq + j) * 32 + kk] = e[j];
        }
    }
#if HAVE_TDM
    if (wave == 0) __builtin_amdgcn_s_wait_tensorcnt(0);
#endif
    __syncthreads();

    for (int k0 = 0; k0 < K; k0 += 32) {
        const int  buf = (k0 >> 5) & 1;
        const bool nxt = (k0 + 32 < K);

        // ---- kick next tile: TDM for A, global loads (regs) for B ----
        uint4 r[2];
        if (nxt) {
#if HAVE_TDM
            if (wave == 0)
                tdm_load_2d(lds_off32(&a_lds[buf ^ 1][0]),
                            A + (size_t)m0 * K + k0 + 32, 128, 32, K);
#endif
#pragma unroll
            for (int i = 0; i < 2; ++i) {
                int idx = tid + i * 256;
                int kk = idx >> 4, nq = (idx & 15) << 3;
                r[i] = *(const uint4*)(Bm + (size_t)(k0 + 32 + kk) * N + n0 + nq);
            }
        }

        // ---- compute from current buffer ----
        const int arow = wave * 16 + lc;
        uint4 a0 = *(const uint4*)(&a_lds[buf][arow * 32 + hl * 8]);
        uint4 a1 = *(const uint4*)(&a_lds[buf][arow * 32 + 16 + hl * 8]);
        v16bf afrag = mk16(a0, a1);
#pragma unroll
        for (int g = 0; g < 8; ++g) {
            const int col = g * 16 + lc;
            uint4 b0 = *(const uint4*)(&b_lds[buf][col * 32 + hl * 8]);
            uint4 b1 = *(const uint4*)(&b_lds[buf][col * 32 + 16 + hl * 8]);
            acc[g] = wmma_bf16(afrag, mk16(b0, b1), acc[g]);
        }

        // ---- finish next-tile staging ----
        if (nxt) {
#if !HAVE_TDM
#pragma unroll
            for (int i = 0; i < 2; ++i) {
                int idx = tid + i * 256;
                int row = idx >> 2, kq = (idx & 3) << 3;
                uint4 d = *(const uint4*)(A + (size_t)(m0 + row) * K + k0 + 32 + kq);
                *(uint4*)(&a_lds[buf ^ 1][row * 32 + kq]) = d;
            }
#endif
#pragma unroll
            for (int i = 0; i < 2; ++i) {
                int idx = tid + i * 256;
                int kk = idx >> 4, nq = (idx & 15) << 3;
                const unsigned short* e = (const unsigned short*)&r[i];
#pragma unroll
                for (int j = 0; j < 8; ++j)
                    b_lds[buf ^ 1][(nq + j) * 32 + kk] = e[j];
            }
#if HAVE_TDM
            if (wave == 0) __builtin_amdgcn_s_wait_tensorcnt(0);
#endif
        }
        __syncthreads();
    }

    // C layout: VGPR r <-> row r (lanes 0-15) / r+8 (lanes 16-31), col = lc
#pragma unroll
    for (int g = 0; g < 8; ++g) {
        const int col = n0 + g * 16 + lc;
#pragma unroll
        for (int r8 = 0; r8 < 8; ++r8) {
            const int row = m0 + wave * 16 + r8 + hl * 8;
            float v = acc[g][r8];
            if (OUT_BF16)
                ((unsigned short*)C)[(size_t)row * N + col] = f32_to_bf16(v);
            else
                ((float*)C)[(size_t)row * N + col] = v;
        }
    }
}

// ---------------------------------------------------------------------------
// Neox RoPE over q,k inside the packed bf16 QKV buffer [tokens, 12288].
// ---------------------------------------------------------------------------
__global__ __launch_bounds__(256) void rope_kernel(
        const int* __restrict__ positions, unsigned short* __restrict__ qkv,
        int S) {
    long idx = (long)blockIdx.x * blockDim.x + threadIdx.x; // tokens*32*64
    int  i   = (int)(idx & 63);
    long th  = idx >> 6;
    int  h   = (int)(th & 31);
    long tok = th >> 5;
    int  s   = (int)(tok % S);
    float pos = (float)positions[s];
    float inv = __expf(-(float)i * (9.210340371976184f / 64.0f)); // 10000^(-i/64)
    float ang = pos * inv;
    float c, sn;
    __sincosf(ang, &c, &sn);

    size_t qb = (size_t)tok * 12288 + (size_t)h * 128 + i;
    size_t kb = qb + 4096;
    {
        float x1 = bf16_to_f32(qkv[qb]);
        float x2 = bf16_to_f32(qkv[qb + 64]);
        qkv[qb]      = f32_to_bf16(x1 * c - x2 * sn);
        qkv[qb + 64] = f32_to_bf16(x2 * c + x1 * sn);
    }
    {
        float x1 = bf16_to_f32(qkv[kb]);
        float x2 = bf16_to_f32(qkv[kb + 64]);
        qkv[kb]      = f32_to_bf16(x1 * c - x2 * sn);
        qkv[kb + 64] = f32_to_bf16(x2 * c + x1 * sn);
    }
}

// ---------------------------------------------------------------------------
// Causal flash attention. Grid (S/128, heads, batch). Block = 8 waves.
// K tile streamed by TDM; V tile register-staged + transposed into LDS.
// Online softmax with per-row stats in lanes 0-15 of each wave.
// ---------------------------------------------------------------------------
__global__ __launch_bounds__(256) void flash_attn_kernel(
        const unsigned short* __restrict__ qkv,  // [B*S, 12288] bf16
        unsigned short* __restrict__ out,        // [B*S, 4096]  bf16
        int S) {
    const int q0   = blockIdx.x * 128;
    const int h    = blockIdx.y;
    const int b    = blockIdx.z;
    const int tid  = threadIdx.x;
    const int lane = tid & 31;
    const int wave = tid >> 5;
    const int hl   = (lane >> 4) & 1;
    const int lc   = lane & 15;

    __shared__ unsigned short k_lds[32 * 128];       // [key][d]
    __shared__ unsigned short vt_lds[128 * 32];      // [d][key] (transposed)
    __shared__ float          s_scr[8][16 * 32];
    __shared__ unsigned short p_scr[8][16 * 32];
    __shared__ float          alpha_lds[8][16];
    __shared__ float          lsum_lds[8][16];

    const size_t tokbase = (size_t)b * S;
    const float  scale   = 0.08838834764831845f;     // 1/sqrt(128)

    // ---- Q fragments (A-matrix layout) straight from global ----
    const int qrow = q0 + wave * 16 + lc;
    const unsigned short* qptr =
        qkv + (tokbase + qrow) * 12288 + (size_t)h * 128;
    v16bf qf[4];
#pragma unroll
    for (int dc = 0; dc < 4; ++dc) {
        uint4 a0 = *(const uint4*)(qptr + dc * 32 + hl * 8);
        uint4 a1 = *(const uint4*)(qptr + dc * 32 + 16 + hl * 8);
        qf[dc] = mk16(a0, a1);
    }

    const v8f vzero = {0.f, 0.f, 0.f, 0.f, 0.f, 0.f, 0.f, 0.f};
    v8f o[8];
#pragma unroll
    for (int g = 0; g < 8; ++g) o[g] = vzero;
    float m_i = -3.0e38f;
    float l_i = 0.0f;

    const int kv_end = q0 + 128;
    for (int kv = 0; kv < kv_end; kv += 32) {
        // ---- stage K via TDM, V via regs + transposed LDS scatter ----
#if HAVE_TDM
        if (wave == 0)
            tdm_load_2d(lds_off32(k_lds),
                        qkv + (tokbase + kv) * 12288 + 4096 + (size_t)h * 128,
                        32, 128, 12288);
#endif
        {
            uint4 vr[2];
#pragma unroll
            for (int i = 0; i < 2; ++i) {
                int idx  = tid + i * 256;       // 512 x uint4
                int krow = idx >> 4;            // key 0..31
                int kdq  = (idx & 15) << 3;     // d offset 0..120
                vr[i] = *(const uint4*)(qkv + (tokbase + kv + krow) * 12288 +
                                        8192 + (size_t)h * 128 + kdq);
#if !HAVE_TDM
                uint4 kd = *(const uint4*)(qkv + (tokbase + kv + krow) * 12288 +
                                           4096 + (size_t)h * 128 + kdq);
                *(uint4*)(k_lds + krow * 128 + kdq) = kd;
#endif
            }
#pragma unroll
            for (int i = 0; i < 2; ++i) {
                int idx  = tid + i * 256;
                int krow = idx >> 4;
                int kdq  = (idx & 15) << 3;
                const unsigned short* e = (const unsigned short*)&vr[i];
#pragma unroll
                for (int j = 0; j < 8; ++j) vt_lds[(kdq + j) * 32 + krow] = e[j];
            }
        }
        // prefetch next kv tile's V rows while this tile computes
        if (kv + 32 < kv_end)
            __builtin_prefetch(qkv + (tokbase + kv + 32 + (tid & 31)) * 12288 +
                               8192 + (size_t)h * 128, 0, 1);
#if HAVE_TDM
        if (wave == 0) __builtin_amdgcn_s_wait_tensorcnt(0);
#endif
        __syncthreads();

        // ---- S = Q K^T : two 16x16 tiles ----
        v8f s0 = vzero, s1 = vzero;
#pragma unroll
        for (int dc = 0; dc < 4; ++dc) {
            uint4 b0 = *(const uint4*)(k_lds + lc * 128 + dc * 32 + hl * 8);
            uint4 b1 = *(const uint4*)(k_lds + lc * 128 + dc * 32 + 16 + hl * 8);
            s0 = wmma_bf16(qf[dc], mk16(b0, b1), s0);
            uint4 c0 = *(const uint4*)(k_lds + (16 + lc) * 128 + dc * 32 + hl * 8);
            uint4 c1 = *(const uint4*)(k_lds + (16 + lc) * 128 + dc * 32 + 16 + hl * 8);
            s1 = wmma_bf16(qf[dc], mk16(c0, c1), s1);
        }

        // ---- scale + causal mask, spill to per-wave LDS scratch ----
#pragma unroll
        for (int r = 0; r < 8; ++r) {
            const int qa = q0 + wave * 16 + r + hl * 8;
            float v0 = s0[r] * scale;
            if (kv + lc > qa) v0 = -3.0e38f;
            s_scr[wave][(r + hl * 8) * 32 + lc] = v0;
            float v1 = s1[r] * scale;
            if (kv + 16 + lc > qa) v1 = -3.0e38f;
            s_scr[wave][(r + hl * 8) * 32 + 16 + lc] = v1;
        }
        wave_lds_sync();

        // ---- online softmax: lane m (0-15) owns row m ----
        if (lane < 16) {
            float* rowp = &s_scr[wave][lane * 32];
            float mx = m_i;
#pragma unroll
            for (int j = 0; j < 32; ++j) mx = fmaxf(mx, rowp[j]);
            float a   = __expf(m_i - mx);
            float sum = 0.f;
#pragma unroll
            for (int j = 0; j < 32; ++j) {
                float p = __expf(rowp[j] - mx);
                sum += p;
                p_scr[wave][lane * 32 + j] = f32_to_bf16(p);
            }
            m_i = mx;
            l_i = l_i * a + sum;
            alpha_lds[wave][lane] = a;
        }
        wave_lds_sync();

        // ---- rescale O accumulators ----
#pragma unroll
        for (int r = 0; r < 8; ++r) {
            float a = alpha_lds[wave][r + hl * 8];
#pragma unroll
            for (int g = 0; g < 8; ++g) o[g][r] *= a;
        }

        // ---- O += P V ----
        uint4 p0 = *(const uint4*)(&p_scr[wave][lc * 32 + hl * 8]);
        uint4 p1 = *(const uint4*)(&p_scr[wave][lc * 32 + 16 + hl * 8]);
        v16bf pf = mk16(p0, p1);
#pragma unroll
        for (int g = 0; g < 8; ++g) {
            uint4 b0 = *(const uint4*)(vt_lds + (g * 16 + lc) * 32 + hl * 8);
            uint4 b1 = *(const uint4*)(vt_lds + (g * 16 + lc) * 32 + 16 + hl * 8);
            o[g] = wmma_bf16(pf, mk16(b0, b1), o[g]);
        }
        __syncthreads();
    }

    // ---- normalize and write O ----
    if (lane < 16) lsum_lds[wave][lane] = l_i;
    wave_lds_sync();
#pragma unroll
    for (int r = 0; r < 8; ++r) {
        const float inv = 1.0f / lsum_lds[wave][r + hl * 8];
        const int row_abs = q0 + wave * 16 + r + hl * 8;
        const size_t ob = (tokbase + row_abs) * 4096 + (size_t)h * 128;
#pragma unroll
        for (int g = 0; g < 8; ++g)
            out[ob + g * 16 + lc] = f32_to_bf16(o[g][r] * inv);
    }
}

// ---------------------------------------------------------------------------
// Launch: cvt -> QKV GEMM -> RoPE -> flash attention -> O-proj GEMM
// ---------------------------------------------------------------------------
extern "C" void kernel_launch(void* const* d_in, const int* in_sizes, int n_in,
                              void* d_out, int out_size, void* d_ws,
                              size_t ws_size, hipStream_t stream) {
    const int*   positions = (const int*)d_in[0];
    const float* hidden    = (const float*)d_in[1];
    const float* w_pack    = (const float*)d_in[2];
    const float* w_o       = (const float*)d_in[3];
    float*       out       = (float*)d_out;

    const int S = 2048, H = 4096, N3 = 12288;
    const int M = 2 * S;               // 4096 tokens

    unsigned short* hs_b = (unsigned short*)d_ws;          // [M,H]
    unsigned short* wp_b = hs_b + (size_t)M * H;           // [H,N3]
    unsigned short* wo_b = wp_b + (size_t)H * N3;          // [H,H]
    unsigned short* qkv  = wo_b + (size_t)H * H;           // [M,N3]
    unsigned short* attn = qkv + (size_t)M * N3;           // [M,H]

    {
        long n1 = (long)M * H, n2 = (long)H * N3, n3 = (long)H * H;
        cvt_f32_bf16_kernel<<<(unsigned)(n1 / 4 / 256), 256, 0, stream>>>(hidden, hs_b, n1);
        cvt_f32_bf16_kernel<<<(unsigned)(n2 / 4 / 256), 256, 0, stream>>>(w_pack, wp_b, n2);
        cvt_f32_bf16_kernel<<<(unsigned)(n3 / 4 / 256), 256, 0, stream>>>(w_o, wo_b, n3);
    }
    gemm_bf16_kernel<true><<<dim3(N3 / 128, M / 128), 256, 0, stream>>>(
        hs_b, wp_b, qkv, M, N3, H);
    {
        long nt = (long)M * 32 * 64;
        rope_kernel<<<(unsigned)(nt / 256), 256, 0, stream>>>(positions, qkv, S);
    }
    flash_attn_kernel<<<dim3(S / 128, 32, 2), 256, 0, stream>>>(qkv, attn, S);
    gemm_bf16_kernel<false><<<dim3(H / 128, M / 128), 256, 0, stream>>>(
        attn, wo_b, out, M, H, H);
}
